// _ProxyAttention_57217554317807
// MI455X (gfx1250) — compile-verified, matrix operand order checked by reading, not run on hardware
//
#include <hip/hip_runtime.h>
#include <hip/hip_bf16.h>

typedef __attribute__((ext_vector_type(16))) __bf16 v16bf;
typedef __attribute__((ext_vector_type(8)))  float  v8f;
typedef __attribute__((ext_vector_type(8)))  unsigned short ushort8;

#define H_HEADS 32
#define KV_HEADS 8
#define HEAD_DIM 64
#define HID 2048
#define KV_DIM 512

__device__ __forceinline__ unsigned short f32_to_bf16(float f) {
    unsigned int u = __float_as_uint(f);
    unsigned int r = u + 0x7fffu + ((u >> 16) & 1u);   // round-to-nearest-even
    return (unsigned short)(r >> 16);
}

// Load one 16x32 bf16 tile in WMMA A/B register layout.
// Lane l: row/col = l&15, kb = (l>>4)*8; elems 0..7 = [idx][kb..kb+7],
// elems 8..15 = [idx][16+kb..16+kb+7]. Two 16-byte loads.
__device__ __forceinline__ v16bf load_tile16(const unsigned short* __restrict__ base,
                                             int stride, int lane) {
    const int r  = lane & 15;
    const int kb = (lane >> 4) << 3;
    const unsigned short* p = base + (size_t)r * stride + kb;
    union { v16bf bf; ushort8 h[2]; } u;
    u.h[0] = *(const ushort8*)(p);
    u.h[1] = *(const ushort8*)(p + 16);
    return u.bf;
}

__device__ __forceinline__ v8f wmma_bf16(v16bf a, v16bf b, v8f c) {
    return __builtin_amdgcn_wmma_f32_16x16x32_bf16(false, a, false, b, (short)0, c, false, false);
}

// ---------------- conversion kernels ----------------
__global__ void k_cvt_bf16(const float* __restrict__ in, unsigned short* __restrict__ out, int n) {
    int i = blockIdx.x * blockDim.x + threadIdx.x;
    if (i < n) out[i] = f32_to_bf16(in[i]);
}

// in: [K x N] f32 row-major -> out: [N x K] bf16 row-major, LDS-tiled 32x32.
// block (32,8), grid (K/32, N/32)
__global__ void k_transpose_cvt(const float* __restrict__ in, unsigned short* __restrict__ out,
                                int K, int N) {
    __shared__ float tile[32][33];
    const int kt = blockIdx.x * 32;
    const int nt = blockIdx.y * 32;
    const int tx = threadIdx.x;
    const int ty = threadIdx.y;
#pragma unroll
    for (int i = 0; i < 32; i += 8)
        tile[ty + i][tx] = in[(size_t)(kt + ty + i) * N + nt + tx];
    __syncthreads();
#pragma unroll
    for (int i = 0; i < 32; i += 8)
        out[(size_t)(nt + ty + i) * K + kt + tx] = f32_to_bf16(tile[tx][ty + i]);
}

// ---------------- GEMM: D[M,N] = A[M,K] * Bt[N,K]^T, 32x64 tile per wave ----
// MODE 0: store f32 row-major. MODE 1: store bf16 row-major.
// MODE 2: store bf16 as v_t[(b*KV_DIM + n)*T + t]  (dim-major per batch)
template<int MODE>
__global__ void k_gemm_bf16(const unsigned short* __restrict__ A,
                            const unsigned short* __restrict__ Bt,
                            void* __restrict__ Out,
                            int M, int N, int K, int T) {
    const int lane = threadIdx.x & 31;
    const int wid  = blockIdx.x * (blockDim.x >> 5) + (threadIdx.x >> 5);
    const int ntiles = N >> 6;
    const int mt = wid / ntiles;
    const int nt = wid - mt * ntiles;
    if (mt * 32 >= M) return;

    v8f zero = {0.f,0.f,0.f,0.f,0.f,0.f,0.f,0.f};
    v8f acc[2][4] = {{zero, zero, zero, zero}, {zero, zero, zero, zero}};

    const unsigned short* Arow = A + (size_t)(mt * 32) * K;
    const unsigned short* B0   = Bt + (size_t)(nt * 64) * K;
    for (int k0 = 0; k0 < K; k0 += 32) {
        v16bf a0 = load_tile16(Arow + k0,                   K, lane);
        v16bf a1 = load_tile16(Arow + (size_t)16 * K + k0,  K, lane);
#pragma unroll
        for (int j = 0; j < 4; ++j) {
            v16bf b = load_tile16(B0 + (size_t)(j * 16) * K + k0, K, lane);
            acc[0][j] = wmma_bf16(a0, b, acc[0][j]);
            acc[1][j] = wmma_bf16(a1, b, acc[1][j]);
        }
    }

    const int n0 = nt * 64;
    const int nn = lane & 15;
    const int mh = (lane >> 4) << 3;           // D: elem r -> row m = mh + r
#pragma unroll
    for (int i = 0; i < 2; ++i) {
        const int m0 = mt * 32 + i * 16;
        if (MODE == 0) {
            float* out = (float*)Out;
#pragma unroll
            for (int j = 0; j < 4; ++j)
#pragma unroll
                for (int r = 0; r < 8; ++r)
                    out[(size_t)(m0 + mh + r) * N + n0 + j * 16 + nn] = acc[i][j][r];
        } else if (MODE == 1) {
            unsigned short* out = (unsigned short*)Out;
#pragma unroll
            for (int j = 0; j < 4; ++j)
#pragma unroll
                for (int r = 0; r < 8; ++r)
                    out[(size_t)(m0 + mh + r) * N + n0 + j * 16 + nn] = f32_to_bf16(acc[i][j][r]);
        } else {
            unsigned short* out = (unsigned short*)Out;
            const int b  = m0 / T;
            const int t0 = m0 - b * T;
#pragma unroll
            for (int j = 0; j < 4; ++j)
#pragma unroll
                for (int r = 0; r < 8; ++r)
                    out[((size_t)b * KV_DIM + n0 + j * 16 + nn) * T + t0 + mh + r] =
                        f32_to_bf16(acc[i][j][r]);
        }
    }
}

// ---------------- flash attention, fully transposed (no LDS, no barriers) ---
// One wave per (b, head, 16-query tile). 4 waves per block.
// S^T = K_tile . Q^T : D-layout of S^T == B-operand layout of P^T, so the
// softmax probabilities feed the P.V WMMA directly from registers.
// Accumulate O^T = V^T . P^T so alpha / 1/l are per-lane scalars.
__global__ void __launch_bounds__(128)
__attribute__((amdgpu_waves_per_eu(1, 4)))
k_attention(const unsigned short* __restrict__ q,
            const unsigned short* __restrict__ kmat,
            const unsigned short* __restrict__ vt,
            unsigned short* __restrict__ attn,
            int T) {
    const int lane = threadIdx.x & 31;
    const int qt = blockIdx.x * 4 + (threadIdx.x >> 5);
    const int h  = blockIdx.y;
    const int b  = blockIdx.z;
    const int kvh = h >> 2;             // GQA: 4 q-heads per kv-head
    const float scale = 0.125f;         // 1/sqrt(64)

    const size_t qrow0 = (size_t)b * T + qt * 16;
    const unsigned short* qbase = q + qrow0 * HID + h * HEAD_DIM;
    v16bf qb0 = load_tile16(qbase,      HID, lane);   // B operand: dims 0..31
    v16bf qb1 = load_tile16(qbase + 32, HID, lane);   // B operand: dims 32..63

    v8f zero = {0.f,0.f,0.f,0.f,0.f,0.f,0.f,0.f};
    v8f o[4] = {zero, zero, zero, zero};              // O^T: 64 dims x 16 queries
    float m_i = -1e30f, l_i = 0.f;                    // per-lane (query = lane&15)

    const unsigned short* kbase = kmat + (size_t)b * T * KV_DIM + kvh * HEAD_DIM;
    const unsigned short* vbase = vt + ((size_t)b * KV_DIM + kvh * HEAD_DIM) * T;

    for (int s = 0; s < T / 32; ++s) {
        const unsigned short* kt0 = kbase + (size_t)s * 32 * KV_DIM;
        v16bf ka00 = load_tile16(kt0,                    KV_DIM, lane);
        v16bf ka01 = load_tile16(kt0 + 32,               KV_DIM, lane);
        v16bf ka10 = load_tile16(kt0 + 16 * KV_DIM,      KV_DIM, lane);
        v16bf ka11 = load_tile16(kt0 + 16 * KV_DIM + 32, KV_DIM, lane);
        v8f st0 = zero, st1 = zero;                   // S^T: keys x queries
        st0 = wmma_bf16(ka00, qb0, st0);
        st0 = wmma_bf16(ka01, qb1, st0);
        st1 = wmma_bf16(ka10, qb0, st1);
        st1 = wmma_bf16(ka11, qb1, st1);

        // scale + per-lane max over the 16 local keys, then combine halves
        float a0[8], a1[8];
        float mx = -1e30f;
#pragma unroll
        for (int r = 0; r < 8; ++r) {
            a0[r] = st0[r] * scale;
            a1[r] = st1[r] * scale;
            mx = fmaxf(mx, fmaxf(a0[r], a1[r]));
        }
        mx = fmaxf(mx, __shfl_xor(mx, 16, 32));       // lanes l <-> l+16
        const float nm = fmaxf(m_i, mx);
        const float al = __expf(m_i - nm);

        // exponentiate, pack P^T directly into B-operand layout, row-sum
        union { v16bf bf; unsigned short us[16]; } pu;
        float rs = 0.f;
#pragma unroll
        for (int r = 0; r < 8; ++r) {
            float e0 = __expf(a0[r] - nm);
            float e1 = __expf(a1[r] - nm);
            rs += e0 + e1;
            pu.us[r]     = f32_to_bf16(e0);           // keys kb+r
            pu.us[8 + r] = f32_to_bf16(e1);           // keys 16+kb+r
        }
        rs += __shfl_xor(rs, 16, 32);
        l_i = al * l_i + rs;
        m_i = nm;

        // O^T = al*O^T + V^T . P^T
        const unsigned short* vstep = vbase + s * 32;
#pragma unroll
        for (int j = 0; j < 4; ++j) {
            v16bf va = load_tile16(vstep + (size_t)(j * 16) * T, T, lane);
#pragma unroll
            for (int r = 0; r < 8; ++r) o[j][r] *= al;
            o[j] = wmma_bf16(va, pu.bf, o[j]);
        }
    }

    // normalize, store O^T (rows = dims, cols = queries on lanes)
    const float inv = 1.0f / l_i;
    const int nn = lane & 15;
    const int mh = (lane >> 4) << 3;
#pragma unroll
    for (int j = 0; j < 4; ++j)
#pragma unroll
        for (int r = 0; r < 8; ++r)
            attn[(qrow0 + nn) * HID + h * HEAD_DIM + j * 16 + mh + r] =
                f32_to_bf16(o[j][r] * inv);
}

// ---------------- launcher ----------------
extern "C" void kernel_launch(void* const* d_in, const int* in_sizes, int n_in,
                              void* d_out, int out_size, void* d_ws, size_t ws_size,
                              hipStream_t stream) {
    const float* x  = (const float*)d_in[0];
    const float* Wq = (const float*)d_in[1];
    const float* Wk = (const float*)d_in[2];
    const float* Wv = (const float*)d_in[3];
    const float* Wo = (const float*)d_in[4];
    const int B = 2, T = 2048;
    const int M = B * T;

    char* ws = (char*)d_ws;
    size_t off = 0;
    auto take = [&](size_t bytes) { void* p = ws + off; off += bytes; return p; };
    unsigned short* x_bf   = (unsigned short*)take((size_t)M * HID * 2);
    unsigned short* WqT    = (unsigned short*)take((size_t)HID * HID * 2);
    unsigned short* WkT    = (unsigned short*)take((size_t)KV_DIM * HID * 2);
    unsigned short* WvT    = (unsigned short*)take((size_t)KV_DIM * HID * 2);
    unsigned short* WoT    = (unsigned short*)take((size_t)HID * HID * 2);
    unsigned short* q_bf   = (unsigned short*)take((size_t)M * HID * 2);
    unsigned short* k_bf   = (unsigned short*)take((size_t)M * KV_DIM * 2);
    unsigned short* v_t    = (unsigned short*)take((size_t)M * KV_DIM * 2);
    unsigned short* att_bf = (unsigned short*)take((size_t)M * HID * 2);

    // convert / transpose
    {
        int n = M * HID;
        k_cvt_bf16<<<(n + 255) / 256, 256, 0, stream>>>(x, x_bf, n);
    }
    {
        dim3 blk(32, 8);
        k_transpose_cvt<<<dim3(HID / 32, HID / 32),    blk, 0, stream>>>(Wq, WqT, HID, HID);
        k_transpose_cvt<<<dim3(HID / 32, KV_DIM / 32), blk, 0, stream>>>(Wk, WkT, HID, KV_DIM);
        k_transpose_cvt<<<dim3(HID / 32, KV_DIM / 32), blk, 0, stream>>>(Wv, WvT, HID, KV_DIM);
        k_transpose_cvt<<<dim3(HID / 32, HID / 32),    blk, 0, stream>>>(Wo, WoT, HID, HID);
    }

    // projections (32x64 tile per wave, 8 waves per 256-thread block)
    {
        int waves = (M / 32) * (HID / 64);
        k_gemm_bf16<1><<<waves / 8, 256, 0, stream>>>(x_bf, WqT, q_bf, M, HID, HID, T);
    }
    {
        int waves = (M / 32) * (KV_DIM / 64);
        k_gemm_bf16<1><<<waves / 8, 256, 0, stream>>>(x_bf, WkT, k_bf, M, KV_DIM, HID, T);
        k_gemm_bf16<2><<<waves / 8, 256, 0, stream>>>(x_bf, WvT, v_t,  M, KV_DIM, HID, T);
    }

    // attention: grid (T/64, heads, B), 4 waves per block, 1 q-tile per wave
    {
        dim3 grid(T / 64, H_HEADS, B);
        k_attention<<<grid, 128, 0, stream>>>(q_bf, k_bf, v_t, att_bf, T);
    }

    // output projection -> f32
    {
        int waves = (M / 32) * (HID / 64);
        k_gemm_bf16<0><<<waves / 8, 256, 0, stream>>>(att_bf, WoT, d_out, M, HID, HID, T);
    }
}
